// TriVec_6476810682566
// MI455X (gfx1250) — matrix operand order repaired and checked by reading, not compiled, not run on hardware
//
#include <hip/hip_runtime.h>
#include <hip/hip_bf16.h>

typedef __attribute__((ext_vector_type(16))) __bf16 v16bf;
typedef __attribute__((ext_vector_type(8)))  float  v8f;
typedef int v4i_vs __attribute__((vector_size(16)));   // matches builtin proto

#define V_TOTAL     500000
#define KDIM        384
#define B_BATCH     256
#define LAMB_F      0.01f
#define MTILES      16            // 256 rows / 16
#define KSTEPS      12            // 384 / 32
#define ITER_COLS   128           // 8 waves * 16 columns
#define CHUNK_ITERS 15
#define CHUNK_COLS  (ITER_COLS * CHUNK_ITERS)          // 1920
#define NCHUNK      ((V_TOTAL + CHUNK_COLS - 1) / CHUNK_COLS)  // 261
#define A_FRAG_BYTES (KSTEPS * MTILES * 32 * 16 * 2)   // 196608 B bf16 fragments
#define KSG         2                                   // k-steps per async group
#define GROUPS      6                                   // groups per iter (12 ks)
#define EBUF_ONE    32768                               // 8 waves * 2 ks * 32 * 64B
#define NBUFS       3                                   // pipeline depth 3
#define SMEM_BYTES  (A_FRAG_BYTES + NBUFS * EBUF_ONE + 256 * 4)   // 295936

// ---- CDNA5 async global->LDS copy (ASYNCcnt-tracked), guarded -------------
#if defined(__has_builtin)
#  if __has_builtin(__builtin_amdgcn_global_load_async_to_lds_b128)
#    define HAVE_ASYNC_LDS 1
#  endif
#endif
#ifndef HAVE_ASYNC_LDS
#  define HAVE_ASYNC_LDS 0
#endif

__device__ static __forceinline__ void async_g2l_b128(const void* g, void* l) {
#if HAVE_ASYNC_LDS
    __builtin_amdgcn_global_load_async_to_lds_b128(
        (__attribute__((address_space(1))) v4i_vs*)g,
        (__attribute__((address_space(3))) v4i_vs*)l, 0, 0);
#else
    *(float4*)l = *(const float4*)g;   // synchronous fallback
#endif
}
__device__ static __forceinline__ void wait_async16() {
    asm volatile("s_wait_asynccnt 16" ::: "memory");
}
__device__ static __forceinline__ void wait_async8() {
    asm volatile("s_wait_asynccnt 8" ::: "memory");
}
__device__ static __forceinline__ void wait_async0() {
    asm volatile("s_wait_asynccnt 0" ::: "memory");
}

// ---------------------------------------------------------------------------
// Kernel 1: gather s/p/o, build plain fp32 A matrices, scores[b], L3 reg.
// ---------------------------------------------------------------------------
__global__ void __launch_bounds__(256) trivec_prep(
    const float* __restrict__ E, const int* __restrict__ sIdx,
    const int* __restrict__ pIdx, const int* __restrict__ oIdx,
    float* __restrict__ Aplain,      // [2][256][384] fp32
    float* __restrict__ out,         // d_out: scores at [0..255]
    float* __restrict__ regOut)      // 1 float
{
    int b = threadIdx.x;
    const float* s = E + (size_t)sIdx[b] * KDIM;
    const float* p = E + (size_t)pIdx[b] * KDIM;
    const float* o = E + (size_t)oIdx[b] * KDIM;
    float* Aobj = Aplain + (size_t)b * KDIM;
    float* Asub = Aplain + (size_t)(B_BATCH + b) * KDIM;

    float score = 0.0f, reg = 0.0f;
    for (int d = 0; d < KDIM; ++d) {
        int i = d >> 7, kk = d & 127;
        float se = s[d], pe = p[d], oe = o[d];
        float orev = o[(2 - i) * 128 + kk];
        Aobj[(2 - i) * 128 + kk] = se * pe;
        Asub[d] = pe * orev;
        score = fmaf(se * pe, orev, score);
        float as = fabsf(se), ap = fabsf(pe), ao = fabsf(oe);
        reg += as * as * as + ap * ap * ap + ao * ao * ao;
    }
    out[b] = score;

    __shared__ float red[256];
    red[b] = reg;
    __syncthreads();
    for (int st = 128; st > 0; st >>= 1) {
        if (b < st) red[b] += red[b + st];
        __syncthreads();
    }
    if (b == 0) regOut[0] = red[0] * (LAMB_F / 3.0f);
}

// ---------------------------------------------------------------------------
// Kernel 2: fused GEMM [256x384]x[384xVchunk] + streaming sum-of-exp per row.
// blockIdx.y = task (0: obj, 1: subj); blockIdx.x = N chunk.
// A: LDS-resident bf16 fragments [ks][mt][lane][16] (immediate DS offsets),
//    consumed with a 1-deep ds_load ping-pong so LDS latency overlaps WMMA.
// E: wave-private async staging, 2 k-steps/group, 3 rotating buffers,
//    always 2 groups in flight (s_wait_asynccnt 16).
// LSE: per-lane running sums in VGPRs; cross-lane reduce once per chunk.
// ---------------------------------------------------------------------------
__global__ void __launch_bounds__(256, 1) trivec_gemm_lse(
    const float* __restrict__ E,        // [V][384] fp32
    const float* __restrict__ Aplain,   // [2][256][384] fp32
    float* __restrict__ partials)       // [2][NCHUNK][256]
{
    extern __shared__ char smem[];
    __bf16* aFrag = (__bf16*)smem;                         // [KS][MT][32][16]
    char*   ebuf  = smem + A_FRAG_BYTES;                   // 3 x 32KB E staging
    float*  ldsS  = (float*)(smem + A_FRAG_BYTES + NBUFS * EBUF_ONE); // [256]

    const int task  = blockIdx.y;
    const int chunk = blockIdx.x;
    const int tid   = threadIdx.x;
    const int lane  = tid & 31;
    const int wave  = tid >> 5;
    const int nlane = lane & 15;   // column within n-tile
    const int h     = lane >> 4;   // lane half

    ldsS[tid] = 0.0f;

    // ---- Phase 1: convert task's A slice into bf16 fragments in LDS ----
    const float* Arow = Aplain + (size_t)task * (B_BATCH * KDIM);
    for (int pr = tid; pr < KSTEPS * MTILES * 32 * 8; pr += 256) {
        int pj = pr & 7;            // VGPR pair index within fragment
        int t  = pr >> 3;
        int ln = t & 31;
        int t2 = t >> 5;
        int mt = t2 & 15;
        int ks = t2 >> 4;
        int m  = mt * 16 + (ln & 15);
        int hh = ln >> 4;
        int kb = 2 * pj + 16 * hh;  // A 16x32 contiguous: k = 16h + e
        int k  = ks * 32 + kb;
        aFrag[pr * 2 + 0] = (__bf16)Arow[m * KDIM + k];
        aFrag[pr * 2 + 1] = (__bf16)Arow[m * KDIM + k + 1];
    }
    __syncthreads();

    // ---- Phase 2: N loop ----
    const int vChunk0 = blockIdx.x * CHUNK_COLS;
    v8f zero = {0.f, 0.f, 0.f, 0.f, 0.f, 0.f, 0.f, 0.f};

    // stage one group (2 k-steps) of this wave's 16 columns into buffer nbuf
    auto stage = [&](int it2, int grp2, int nbuf) {
        int vv = vChunk0 + it2 * ITER_COLS + wave * 16 + nlane;
        if (vv >= V_TOTAL) vv = V_TOTAL - 1;
        const float* srow = E + (size_t)vv * KDIM + h * 16;   // B-frag: k = 16h + e
        char* dstBase = ebuf + nbuf * EBUF_ONE + wave * 4096 + lane * 64;
#pragma unroll
        for (int ks2 = 0; ks2 < KSG; ++ks2) {
#pragma unroll
            for (int part = 0; part < 4; ++part) {
                async_g2l_b128(srow + (grp2 * KSG + ks2) * 32 + part * 4,
                               dstBase + ks2 * 2048 + part * 16);
            }
        }
    };

    stage(0, 0, 0);   // prologue: groups 0 and 1 in flight
    stage(0, 1, 1);

    v8f sums[MTILES];  // per-lane running sum(exp) [row = mt*16 + j + 8h]
#pragma unroll
    for (int mt = 0; mt < MTILES; ++mt) sums[mt] = zero;

    for (int it = 0; it < CHUNK_ITERS; ++it) {
        int v = vChunk0 + it * ITER_COLS + wave * 16 + nlane;

        v8f acc[MTILES];
#pragma unroll
        for (int mt = 0; mt < MTILES; ++mt) acc[mt] = zero;

#pragma unroll
        for (int grp = 0; grp < GROUPS; ++grp) {
            // keep 2 groups in flight: stage group G+2
            int  a2   = grp + 2;
            int  nit2 = it + (a2 >= GROUPS ? 1 : 0);
            int  a1   = grp + 1;
            int  nit1 = it + (a1 >= GROUPS ? 1 : 0);
            bool has2 = (nit2 < CHUNK_ITERS);
            bool has1 = (nit1 < CHUNK_ITERS);
            if (has2)      { stage(nit2, a2 % GROUPS, (grp + 2) % NBUFS); wait_async16(); }
            else if (has1) { wait_async8(); }
            else           { wait_async0(); }

            const char* cbase = ebuf + (grp % NBUFS) * EBUF_ONE + wave * 4096 + lane * 64;
#pragma unroll
            for (int ks2 = 0; ks2 < KSG; ++ks2) {
                int ks = grp * KSG + ks2;
                const float4* c4 = (const float4*)(cbase + ks2 * 2048);
                float4 b0 = c4[0], b1 = c4[1], b2 = c4[2], b3 = c4[3];
                v16bf bfr;
                bfr[0]  = (__bf16)b0.x; bfr[1]  = (__bf16)b0.y;
                bfr[2]  = (__bf16)b0.z; bfr[3]  = (__bf16)b0.w;
                bfr[4]  = (__bf16)b1.x; bfr[5]  = (__bf16)b1.y;
                bfr[6]  = (__bf16)b1.z; bfr[7]  = (__bf16)b1.w;
                bfr[8]  = (__bf16)b2.x; bfr[9]  = (__bf16)b2.y;
                bfr[10] = (__bf16)b2.z; bfr[11] = (__bf16)b2.w;
                bfr[12] = (__bf16)b3.x; bfr[13] = (__bf16)b3.y;
                bfr[14] = (__bf16)b3.z; bfr[15] = (__bf16)b3.w;

                const __bf16* abase = aFrag + ((size_t)ks * MTILES * 32 + lane) * 16;
                v16bf afCur = *(const v16bf*)(abase);
#pragma unroll
                for (int mt = 0; mt < MTILES; ++mt) {
                    // ping-pong: fetch next fragment before this WMMA
                    v16bf afNext = (mt < MTILES - 1)
                                 ? *(const v16bf*)(abase + (mt + 1) * 512)
                                 : afCur;
                    acc[mt] = __builtin_amdgcn_wmma_f32_16x16x32_bf16(
                        false, afCur, false, bfr, (short)0, acc[mt], false, false);
                    afCur = afNext;
                }
            }
        }

        // ---- per-iter epilogue: exp + add into per-lane running sums ----
        bool valid = (v < V_TOTAL);
#pragma unroll
        for (int mt = 0; mt < MTILES; ++mt) {
#pragma unroll
            for (int j = 0; j < 8; ++j) {
                float ex = valid ? __expf(acc[mt][j]) : 0.0f;
                sums[mt][j] += ex;
            }
        }
    }

    // ---- chunk-end reduce: 16-lane butterfly + one LDS atomic per row ----
#pragma unroll
    for (int mt = 0; mt < MTILES; ++mt) {
#pragma unroll
        for (int j = 0; j < 8; ++j) {
            float ex = sums[mt][j];
            ex += __shfl_xor(ex, 1, 32);
            ex += __shfl_xor(ex, 2, 32);
            ex += __shfl_xor(ex, 4, 32);
            ex += __shfl_xor(ex, 8, 32);
            if (nlane == 0) {
                int m = mt * 16 + j + 8 * h;   // C layout: VGPR j -> row j + 8h
                atomicAdd(&ldsS[m], ex);       // ds_add_f32
            }
        }
    }
    __syncthreads();
    partials[((size_t)task * NCHUNK + chunk) * 256 + tid] = ldsS[tid];
}

// ---------------------------------------------------------------------------
// Kernel 3: reduce chunk partials, masked-LSE via S - exp(score), total loss.
// ---------------------------------------------------------------------------
__global__ void __launch_bounds__(256) trivec_finalize(
    const float* __restrict__ partials, const float* __restrict__ regIn,
    float* __restrict__ out)
{
    int b = threadIdx.x;
    float Sobj = 0.0f, Ssub = 0.0f;
    for (int c = 0; c < NCHUNK; ++c) {
        Sobj += partials[(size_t)c * 256 + b];
        Ssub += partials[((size_t)NCHUNK + c) * 256 + b];
    }
    float sc = out[b];              // scores written by prep
    float e  = __expf(sc);
    float lb = -2.0f * sc + logf(Sobj - e) + logf(Ssub - e);

    __shared__ float red[256];
    red[b] = lb;
    __syncthreads();
    for (int st = 128; st > 0; st >>= 1) {
        if (b < st) red[b] += red[b + st];
        __syncthreads();
    }
    if (b == 0) out[256] = red[0] + regIn[0];
}

// ---------------------------------------------------------------------------
extern "C" void kernel_launch(void* const* d_in, const int* in_sizes, int n_in,
                              void* d_out, int out_size, void* d_ws, size_t ws_size,
                              hipStream_t stream) {
    const float* E    = (const float*)d_in[0];
    const int*   sIdx = (const int*)d_in[1];
    const int*   pIdx = (const int*)d_in[2];
    const int*   oIdx = (const int*)d_in[3];
    float* out = (float*)d_out;

    char* ws = (char*)d_ws;
    float* Aplain   = (float*)ws;                                    // 786432 B
    float* partials = (float*)(ws + (size_t)2 * B_BATCH * KDIM * 4); // 2*NCHUNK*256*4
    float* regv     = (float*)(ws + (size_t)2 * B_BATCH * KDIM * 4
                                  + (size_t)2 * NCHUNK * 256 * 4);

    (void)hipFuncSetAttribute((const void*)trivec_gemm_lse,
                              hipFuncAttributeMaxDynamicSharedMemorySize,
                              (int)SMEM_BYTES);

    trivec_prep<<<1, 256, 0, stream>>>(E, sIdx, pIdx, oIdx, Aplain, out, regv);
    trivec_gemm_lse<<<dim3(NCHUNK, 2), 256, SMEM_BYTES, stream>>>(E, Aplain, partials);
    trivec_finalize<<<1, 256, 0, stream>>>(partials, regv, out);
}
